// SymInfoNCELoss_6253472383014
// MI455X (gfx1250) — matrix operand order repaired and checked by reading, not compiled, not run on hardware
//
#include <hip/hip_runtime.h>
#include <hip/hip_bf16.h>
#include <math.h>

// ---------------------------------------------------------------------------
// Symmetric InfoNCE loss, fused for MI455X (gfx1250, wave32, WMMA + TDM).
//   zn = normalize(z)  (f32 -> f16, bounded in [-1,1])
//   For 6 view pairs: S = zn_i @ zn_j^T / 0.1 ; accumulate sum(exp(S)) per
//   row and per column (logits bounded by 10 -> no max subtraction needed),
//   and store diag(S). Final: mean(log(rowsum)+log(colsum)-2*diag)/(2*6).
// Global->LDS staging uses the Tensor Data Mover (tensor_load_to_lds) with
// hardware row padding, double-buffered on TENSORcnt.
// ---------------------------------------------------------------------------

typedef __attribute__((ext_vector_type(16))) _Float16     v16h;
typedef __attribute__((ext_vector_type(8)))  float        v8f;
typedef __attribute__((ext_vector_type(4)))  unsigned int v4u;
typedef __attribute__((ext_vector_type(8)))  int          v8i;
typedef __attribute__((ext_vector_type(4)))  int          v4i;

#define B_ROWS   4096
#define DDIM     256
#define NPAIR    6
#define INV_TEMP 10.0f
#define KSTAGE   64        // K depth per LDS stage
#define LDS_LD   72        // 64-half row + 8-half HW pad (pad_interval=32DW, pad_amount=4DW)

union V16HU { v16h v; uint4 u[2]; };

// ---------------------------------------------------------------------------
// TDM 2-D tile load: global (row-major, row_stride units) -> LDS with
// hardware padding of 16B after every 128B row -> effective LDS_LD=72 halves.
// D# layout per CDNA5 ISA sect. 8.3/8.4.
// This toolchain exposes the 6-arg builtin form:
//   (uint32x4 g0, int32x8 g1, int32x4 g2, int32x4 g3, int32x8, i32 cpol)
__device__ __forceinline__ void tdm_load_tile_f16(unsigned int lds_addr,
                                                  const _Float16* gptr,
                                                  unsigned int tile_w,     // units (halves)
                                                  unsigned int tile_h,
                                                  unsigned int tensor_w,
                                                  unsigned int tensor_h,
                                                  unsigned int row_stride) // units
{
    unsigned long long ga = (unsigned long long)(uintptr_t)gptr;

    v4u g0;
    g0.x = 1u;                                        // count=1, user descriptor
    g0.y = lds_addr;                                  // LDS byte address
    g0.z = (unsigned int)(ga & 0xFFFFFFFFu);          // global_addr[31:0]
    g0.w = (unsigned int)((ga >> 32) & 0x01FFFFFFu)   // global_addr[56:32]
         | (2u << 30);                                // type=2 ("image")

    unsigned int d0 = (1u << 16)                      // data_size = 2 bytes
                    | (1u << 20)                      // pad_enable
                    | (4u << 22)                      // pad_interval: 32 DWORDs (128 B)
                    | (3u << 25);                     // pad_amount:   4 DWORDs (16 B)
    unsigned int d1 = (tensor_w & 0xFFFFu) << 16;                         // dim0[15:0]
    unsigned int d2 = (tensor_w >> 16) | ((tensor_h & 0xFFFFu) << 16);    // dim0 hi | dim1 lo
    unsigned int d3 = (tensor_h >> 16) | ((tile_w & 0xFFFFu) << 16);      // dim1 hi | tile_dim0
    unsigned int d4 = tile_h & 0xFFFFu;                                   // tile_dim1, tile_dim2=0
    unsigned int d5 = row_stride;                                         // dim0_stride[31:0]
    v8i g1 = { (int)d0, (int)d1, (int)d2, (int)d3, (int)d4, (int)d5, 0, 0 };

    v4i gz4 = { 0, 0, 0, 0 };                         // 2-D tensor: groups 2/3 unused
    v8i gz8 = { 0, 0, 0, 0, 0, 0, 0, 0 };
    __builtin_amdgcn_tensor_load_to_lds(g0, g1, gz4, gz4, gz8, 0);
}

// ---------------------------------------------------------------------------
__global__ void zero_kernel(float* __restrict__ p, int n) {
    int i = blockIdx.x * blockDim.x + threadIdx.x;
    if (i < n) p[i] = 0.0f;
}

// ---------------------------------------------------------------------------
// One wave32 per row: sum of squares -> shfl reduce -> 1/max(norm,eps) ->
// f16 row written to workspace [4][4096][256].
__global__ __launch_bounds__(256)
void normalize_f16_kernel(const float* __restrict__ z0,
                          const float* __restrict__ z1,
                          const float* __restrict__ z2,
                          const float* __restrict__ z3,
                          _Float16* __restrict__ zn) {
    int wave = threadIdx.x >> 5;
    int lane = threadIdx.x & 31;
    int row  = blockIdx.x * 8 + wave;          // 0..16383
    int view = row >> 12;
    int r    = row & (B_ROWS - 1);

    const float* src = (view == 0) ? z0 : (view == 1) ? z1 : (view == 2) ? z2 : z3;
    src += r * DDIM + lane * 8;

    float4 a = *(const float4*)(src);
    float4 b = *(const float4*)(src + 4);
    float ss = a.x*a.x + a.y*a.y + a.z*a.z + a.w*a.w
             + b.x*b.x + b.y*b.y + b.z*b.z + b.w*b.w;
    #pragma unroll
    for (int m = 16; m >= 1; m >>= 1) ss += __shfl_xor(ss, m, 32);

    float inv = 1.0f / fmaxf(sqrtf(ss), 1e-8f);

    union { uint4 u; _Float16 h[8]; } o;
    o.h[0] = (_Float16)(a.x * inv); o.h[1] = (_Float16)(a.y * inv);
    o.h[2] = (_Float16)(a.z * inv); o.h[3] = (_Float16)(a.w * inv);
    o.h[4] = (_Float16)(b.x * inv); o.h[5] = (_Float16)(b.y * inv);
    o.h[6] = (_Float16)(b.z * inv); o.h[7] = (_Float16)(b.w * inv);
    *(uint4*)(zn + row * DDIM + lane * 8) = o.u;
}

// ---------------------------------------------------------------------------
// Fused pair GEMM + softmax statistics.
// grid = (4096/128, 4096/128, 6); block = 256 threads = 8 waves.
// Block tile 128x128. K staged 64-deep through LDS via TDM, double-buffered:
// wave 0 issues tensor_load_to_lds for stage s+1, waits TENSORcnt<=2 so stage
// s is resident, then the workgroup barrier releases compute. Each wave owns
// a 64x32 patch = 4x2 WMMA 16x16 tiles of f32 accumulators.
__global__ __launch_bounds__(256)
void pair_gemm_lse_kernel(const _Float16* __restrict__ zn,
                          float* __restrict__ row_sum,
                          float* __restrict__ col_sum,
                          float* __restrict__ diag) {
    __shared__ __align__(16) _Float16 As[2][128 * LDS_LD];
    __shared__ __align__(16) _Float16 Bs[2][128 * LDS_LD];
    __shared__ float rowAcc[128];
    __shared__ float colAcc[128];

    const int pi[NPAIR] = {0, 0, 0, 1, 1, 2};
    const int pj[NPAIR] = {1, 2, 3, 2, 3, 3};
    const int p  = blockIdx.z;
    const int bn = blockIdx.x;
    const int bm = blockIdx.y;

    const _Float16* __restrict__ Ag = zn + pi[p] * B_ROWS * DDIM + bm * 128 * DDIM;
    const _Float16* __restrict__ Bg = zn + pj[p] * B_ROWS * DDIM + bn * 128 * DDIM;

    const int tid    = threadIdx.x;
    const int waveId = tid >> 5;
    const int lane   = tid & 31;
    const int wm = waveId >> 2;    // 0..1 : 64-row group
    const int wn = waveId & 3;     // 0..3 : 32-col group
    const int lh = lane >> 4;      // lane half (0/1)
    const int nn = lane & 15;

    if (tid < 128) rowAcc[tid] = 0.0f;
    else           colAcc[tid - 128] = 0.0f;

    v8f acc[4][2];
    #pragma unroll
    for (int tm = 0; tm < 4; ++tm)
        #pragma unroll
        for (int tn = 0; tn < 2; ++tn) { v8f z = {}; acc[tm][tn] = z; }

    const bool issuer = (waveId == 0);     // wave-uniform; TDM ignores EXEC
    const int NSTAGE  = DDIM / KSTAGE;     // 4

    if (issuer) {
        tdm_load_tile_f16((unsigned int)(uintptr_t)&As[0][0], Ag,
                          KSTAGE, 128, DDIM, B_ROWS, DDIM);
        tdm_load_tile_f16((unsigned int)(uintptr_t)&Bs[0][0], Bg,
                          KSTAGE, 128, DDIM, B_ROWS, DDIM);
    }

    for (int ks = 0; ks < NSTAGE; ++ks) {
        const int cur = ks & 1;
        if (issuer) {
            if (ks + 1 < NSTAGE) {
                const int nxt = (ks + 1) & 1;
                tdm_load_tile_f16((unsigned int)(uintptr_t)&As[nxt][0],
                                  Ag + (ks + 1) * KSTAGE,
                                  KSTAGE, 128, DDIM, B_ROWS, DDIM);
                tdm_load_tile_f16((unsigned int)(uintptr_t)&Bs[nxt][0],
                                  Bg + (ks + 1) * KSTAGE,
                                  KSTAGE, 128, DDIM, B_ROWS, DDIM);
                __builtin_amdgcn_s_wait_tensorcnt((short)2);   // stage ks resident
            } else {
                __builtin_amdgcn_s_wait_tensorcnt((short)0);
            }
        }
        __syncthreads();   // stage ks visible to all waves

        #pragma unroll
        for (int kk = 0; kk < 2; ++kk) {
            const int k0 = kk * 32;
            // A fragment: 16x32 f16 (lane<16: K 0-7/16-23, lane>=16: K 8-15/24-31)
            V16HU af[4];
            #pragma unroll
            for (int tm = 0; tm < 4; ++tm) {
                const _Float16* pa = &As[cur][(wm * 64 + tm * 16 + nn) * LDS_LD + k0 + lh * 8];
                af[tm].u[0] = *(const uint4*)(pa);
                af[tm].u[1] = *(const uint4*)(pa + 16);
            }
            // B fragment: 32x16 f16 (lane<16: K 0-7/16-23, lane>=16: K 8-15/24-31)
            V16HU bf[2];
            #pragma unroll
            for (int tn = 0; tn < 2; ++tn) {
                const _Float16* pb = &Bs[cur][(wn * 32 + tn * 16 + nn) * LDS_LD + k0 + lh * 8];
                bf[tn].u[0] = *(const uint4*)(pb);
                bf[tn].u[1] = *(const uint4*)(pb + 16);
            }
            #pragma unroll
            for (int tm = 0; tm < 4; ++tm)
                #pragma unroll
                for (int tn = 0; tn < 2; ++tn)
                    acc[tm][tn] = __builtin_amdgcn_wmma_f32_16x16x32_f16(
                        false, af[tm].v, false, bf[tn].v,
                        (short)0, acc[tm][tn], false, false);
        }
        __syncthreads();   // compute ks done; buf[ks&1] reusable for stage ks+2
    }

    // Epilogue: exp, row/col partial reductions, diagonal extraction.
    const bool diagBlk = (bm == bn);
    float colPart0 = 0.0f, colPart1 = 0.0f;

    #pragma unroll
    for (int tm = 0; tm < 4; ++tm) {
        float rowv[8];
        #pragma unroll
        for (int v = 0; v < 8; ++v) rowv[v] = 0.0f;

        #pragma unroll
        for (int tn = 0; tn < 2; ++tn) {
            #pragma unroll
            for (int v = 0; v < 8; ++v) {
                float s = acc[tm][tn][v] * INV_TEMP;   // logits bounded by +-10
                float e = __expf(s);
                rowv[v] += e;
                if (tn == 0) colPart0 += e; else colPart1 += e;
                if (diagBlk) {
                    int rloc = wm * 64 + tm * 16 + v + lh * 8;
                    int cloc = wn * 32 + tn * 16 + nn;
                    if (rloc == cloc)
                        diag[p * B_ROWS + bm * 128 + rloc] = s;
                }
            }
        }
        // reduce each row's 32 columns: tn pair summed + width-16 shfl tree
        #pragma unroll
        for (int v = 0; v < 8; ++v) {
            float rs = rowv[v];
            rs += __shfl_xor(rs, 1, 16);
            rs += __shfl_xor(rs, 2, 16);
            rs += __shfl_xor(rs, 4, 16);
            rs += __shfl_xor(rs, 8, 16);
            if (nn == 0)  // lanes 0 (rows v) and 16 (rows v+8)
                atomicAdd(&rowAcc[wm * 64 + tm * 16 + v + lh * 8], rs);
        }
    }
    atomicAdd(&colAcc[wn * 32 + nn], colPart0);
    atomicAdd(&colAcc[wn * 32 + 16 + nn], colPart1);
    __syncthreads();

    if (tid < 128) atomicAdd(&row_sum[p * B_ROWS + bm * 128 + tid], rowAcc[tid]);
    else           atomicAdd(&col_sum[p * B_ROWS + bn * 128 + (tid - 128)], colAcc[tid - 128]);
}

// ---------------------------------------------------------------------------
__global__ __launch_bounds__(256)
void finalize_kernel(const float* __restrict__ row_sum,
                     const float* __restrict__ col_sum,
                     const float* __restrict__ diag,
                     float* __restrict__ out) {
    __shared__ float red[256];
    float acc = 0.0f;
    for (int i = threadIdx.x; i < NPAIR * B_ROWS; i += 256)
        acc += logf(row_sum[i]) + logf(col_sum[i]) - 2.0f * diag[i];
    red[threadIdx.x] = acc;
    __syncthreads();
    #pragma unroll
    for (int s = 128; s > 0; s >>= 1) {
        if (threadIdx.x < s) red[threadIdx.x] += red[threadIdx.x + s];
        __syncthreads();
    }
    if (threadIdx.x == 0)
        out[0] = red[0] / (2.0f * NPAIR * B_ROWS);
}

// ---------------------------------------------------------------------------
extern "C" void kernel_launch(void* const* d_in, const int* in_sizes, int n_in,
                              void* d_out, int out_size, void* d_ws, size_t ws_size,
                              hipStream_t stream) {
    const float* z0 = (const float*)d_in[0];
    const float* z1 = (const float*)d_in[1];
    const float* z2 = (const float*)d_in[2];
    const float* z3 = (const float*)d_in[3];
    float* out = (float*)d_out;

    char* ws = (char*)d_ws;
    _Float16* zn    = (_Float16*)ws;                                 // 4*4096*256*2 = 8 MB
    float* row_sum  = (float*)(ws + (size_t)4 * B_ROWS * DDIM * 2);  // 6*4096 f32
    float* col_sum  = row_sum + NPAIR * B_ROWS;
    float* diag     = col_sum + NPAIR * B_ROWS;

    // row_sum and col_sum are contiguous: one zeroing pass
    zero_kernel<<<(2 * NPAIR * B_ROWS + 255) / 256, 256, 0, stream>>>(row_sum, 2 * NPAIR * B_ROWS);

    normalize_f16_kernel<<<(4 * B_ROWS) / 8, 256, 0, stream>>>(z0, z1, z2, z3, zn);

    dim3 grid(B_ROWS / 128, B_ROWS / 128, NPAIR);
    pair_gemm_lse_kernel<<<grid, 256, 0, stream>>>(zn, row_sum, col_sum, diag);

    finalize_kernel<<<1, 256, 0, stream>>>(row_sum, col_sum, diag, out);
}
